// CrossAttention_19146964205881
// MI455X (gfx1250) — matrix operand order repaired and checked by reading, not compiled, not run on hardware
//
#include <hip/hip_runtime.h>
#include <hip/hip_bf16.h>

// ---------------------------------------------------------------------------
// CrossAttention block for MI455X (gfx1250, wave32, WMMA f32_16x16x32_f16)
//   B=16, NP=1024, NH=512, D=768, H=12, DK=64, F=4D=3072
// 128x128 block tiles (8 WMMA/wave/K-step), double-buffered LDS with one
// barrier per K-step, async global->LDS (GLOBAL_LOAD_ASYNC_TO_LDS_B128,
// ASYNCcnt) with sync register-staged fallback.
// ---------------------------------------------------------------------------

typedef _Float16 h16;
typedef __attribute__((ext_vector_type(16))) _Float16 hv16;
typedef __attribute__((ext_vector_type(8)))  _Float16 hv8;
typedef __attribute__((ext_vector_type(8)))  float    fv8;
typedef __attribute__((ext_vector_type(4)))  float    fv4;
typedef __attribute__((ext_vector_type(4)))  int      iv4;

#define LSA 40   // LDS row stride in halfs (80B -> 16B-aligned rows, de-banked)
#define BM 128
#define BN 128
#define BK 32
#define TILE_H (BM * LSA)   // halfs per (A or B) tile buffer

#if __has_builtin(__builtin_amdgcn_global_load_async_to_lds_b128)
#define HAVE_ASYNC_LDS 1
__device__ __forceinline__ void async_cp16(const h16* g, h16* l) {
  // signature (from hipcc diagnostic): (int4 AS1*, int4 AS3*, imm offset, imm cpol)
  __builtin_amdgcn_global_load_async_to_lds_b128(
      (__attribute__((address_space(1))) iv4*)g,
      (__attribute__((address_space(3))) iv4*)l, 0, 0);
}
__device__ __forceinline__ void wait_async0() {
#if __has_builtin(__builtin_amdgcn_s_wait_asynccnt)
  __builtin_amdgcn_s_wait_asynccnt(0);
#else
  asm volatile("s_wait_asynccnt 0" ::: "memory");
#endif
}
#else
#define HAVE_ASYNC_LDS 0
#endif

// ---- WMMA fragment helpers (CDNA5 ISA 7.12.2 layouts, wave32) --------------

// A fragment: 16x32 f16. lanes 0-15: row M=lane, halfs = K 0..7,16..23
//             lanes 16-31: row M=lane-16, halfs = K 8..15,24..31
__device__ __forceinline__ hv16 frag_a(const h16* S, int m0, int lane) {
  const h16* p = S + (m0 + (lane & 15)) * LSA;
  const int kg = (lane >> 4) * 8;
  hv8 lo = *(const hv8*)(p + kg);
  hv8 hi = *(const hv8*)(p + 16 + kg);
  return __builtin_shufflevector(lo, hi, 0,1,2,3,4,5,6,7,8,9,10,11,12,13,14,15);
}

// B fragment: 32x16 f16, LDS tile stored [N][K]. lane = column N (mod 16),
// lanes 0-15: K 0..7,16..23; lanes 16-31: K 8..15,24..31 (matches frag_a).
__device__ __forceinline__ hv16 frag_b(const h16* S, int n0, int lane) {
  const h16* p = S + (n0 + (lane & 15)) * LSA + (lane >> 4) * 8;
  hv8 lo = *(const hv8*)(p);
  hv8 hi = *(const hv8*)(p + 16);
  return __builtin_shufflevector(lo, hi, 0,1,2,3,4,5,6,7,8,9,10,11,12,13,14,15);
}

__device__ __forceinline__ fv8 wmma16(hv16 a, hv16 b, fv8 c) {
  return __builtin_amdgcn_wmma_f32_16x16x32_f16(false, a, false, b, (short)0, c,
                                                false, false);
}

__device__ __forceinline__ float gelu_exact(float x) {
  return 0.5f * x * (1.0f + erff(x * 0.70710678118654752f));
}

// ---------------------------------------------------------------------------
// Core 128x128x32 double-buffered WMMA loop body, shared by GEMM and logits.
// As/Bs are [2][TILE_H]; computes acc[2][4] (wave tile 32 rows x 64 cols).
// ---------------------------------------------------------------------------
#define WMMA_TILE_LOOP(ABASE, ALD, BBASE, BLD, KTOT)                           \
  {                                                                            \
    /* prologue: tile 0 into buffer 0 */                                       \
    _Pragma("unroll") for (int it = 0; it < 2; ++it) {                         \
      const int slot = t + it * 256;                                           \
      const int row = slot >> 2, seg = (slot & 3) * 8;                         \
      COPY16((ABASE) + (bm + row) * (long)(ALD) + seg, As[0] + row * LSA + seg); \
      COPY16((BBASE) + (bn + row) * (long)(BLD) + seg, Bs[0] + row * LSA + seg); \
    }                                                                          \
    COPY_FENCE();                                                              \
    __syncthreads();                                                           \
    int cur = 0;                                                               \
    for (int k0 = 0; k0 < (KTOT); k0 += BK) {                                  \
      const int nxt = cur ^ 1;                                                 \
      const bool more = (k0 + BK) < (KTOT);                                    \
      STAGE_DECL();                                                            \
      if (more) {                                                              \
        _Pragma("unroll") for (int it = 0; it < 2; ++it) {                     \
          const int slot = t + it * 256;                                       \
          const int row = slot >> 2, seg = (slot & 3) * 8;                     \
          STAGE_IN(it, (ABASE) + (bm + row) * (long)(ALD) + k0 + BK + seg,     \
                       (BBASE) + (bn + row) * (long)(BLD) + k0 + BK + seg,     \
                       As[nxt] + row * LSA + seg, Bs[nxt] + row * LSA + seg);  \
        }                                                                      \
      }                                                                        \
      const hv16 a0 = frag_a(As[cur], m0, lane);                               \
      const hv16 a1 = frag_a(As[cur], m0 + 16, lane);                          \
      _Pragma("unroll") for (int in = 0; in < 4; ++in) {                       \
        const hv16 bfr = frag_b(Bs[cur], n0 + in * 16, lane);                  \
        acc[0][in] = wmma16(a0, bfr, acc[0][in]);                              \
        acc[1][in] = wmma16(a1, bfr, acc[1][in]);                              \
      }                                                                        \
      if (more) {                                                              \
        _Pragma("unroll") for (int it = 0; it < 2; ++it) {                     \
          const int slot = t + it * 256;                                       \
          const int row = slot >> 2, seg = (slot & 3) * 8;                     \
          STAGE_OUT(it, As[nxt] + row * LSA + seg, Bs[nxt] + row * LSA + seg); \
        }                                                                      \
        COPY_FENCE();                                                          \
      }                                                                        \
      __syncthreads();                                                         \
      cur = nxt;                                                               \
    }                                                                          \
  }

#if HAVE_ASYNC_LDS
// direct global -> LDS, no VGPR staging; tracked by ASYNCcnt
#define COPY16(G, L)  async_cp16((G), (L))
#define COPY_FENCE()  wait_async0()
#define STAGE_DECL()
#define STAGE_IN(it, GA, GB, LA, LB)  do { async_cp16((GA), (LA)); async_cp16((GB), (LB)); } while (0)
#define STAGE_OUT(it, LA, LB)
#else
// synchronous fallback: global -> VGPR -> LDS, overlapped across the K-step
#define COPY16(G, L)  (*(hv8*)(L) = *(const hv8*)(G))
#define COPY_FENCE()
#define STAGE_DECL()  hv8 ra[2], rb[2]
#define STAGE_IN(it, GA, GB, LA, LB)  do { ra[it] = *(const hv8*)(GA); rb[it] = *(const hv8*)(GB); } while (0)
#define STAGE_OUT(it, LA, LB)         do { *(hv8*)(LA) = ra[it]; *(hv8*)(LB) = rb[it]; } while (0)
#endif

// ---------------------------------------------------------------------------
// Generic GEMM: C[M,N] = A[M,K](f16) * B[N,K]^T(f16)  (torch Linear layout)
// Epilogue: +bias[col], +resid[row,col], optional exact-GELU;
// writes f32 (outF) and/or f16 (outH). All dims divide tiles exactly.
// grid = (N/128, M/128), block = 256 (8 waves, each owns 32x64).
// ---------------------------------------------------------------------------
__global__ __launch_bounds__(256) void gemm_nt_f16(
    const h16* __restrict__ A, int lda,
    const h16* __restrict__ B, int ldb,
    float* __restrict__ outF, h16* __restrict__ outH, int ldc,
    const float* __restrict__ bias, const float* __restrict__ resid,
    int K, int act)
{
  __shared__ h16 As[2][TILE_H];
  __shared__ h16 Bs[2][TILE_H];
  const int t    = threadIdx.x;
  const int lane = t & 31;
  const int wave = t >> 5;
  const int m0   = (wave & 3) * 32;
  const int n0   = (wave >> 2) * 64;
  const long bm  = (long)blockIdx.y * BM;
  const long bn  = (long)blockIdx.x * BN;

  fv8 acc[2][4] = {};
  WMMA_TILE_LOOP(A, lda, B, ldb, K)

  // Epilogue. C/D layout: VGPR v -> row v (+8 for upper lane half), col = lane&15.
  const int rsub = (lane >> 4) * 8;
  const int csub = lane & 15;
#pragma unroll
  for (int im = 0; im < 2; ++im) {
#pragma unroll
    for (int in = 0; in < 4; ++in) {
      const long row0 = bm + m0 + im * 16 + rsub;
      const long col  = bn + n0 + in * 16 + csub;
      const float bv  = bias ? bias[col] : 0.0f;
#pragma unroll
      for (int v = 0; v < 8; ++v) {
        const long r = row0 + v;
        float val = acc[im][in][v] + bv;
        if (resid) val += resid[r * (long)ldc + col];
        if (act)   val = gelu_exact(val);
        if (outF)  outF[r * (long)ldc + col] = val;
        if (outH)  outH[r * (long)ldc + col] = (h16)val;
      }
    }
  }
}

// ---------------------------------------------------------------------------
// Attention logits: per (b,h): S[1024,512] = Q_bh @ K_bh^T * (exp(tau)/8),
// masked on query rows. grid = (512/128, 1024/128, B*H), block = 256.
// ---------------------------------------------------------------------------
__global__ __launch_bounds__(256) void attn_logits(
    const h16* __restrict__ Q16, const h16* __restrict__ K16,
    const int* __restrict__ mask, const float* __restrict__ tau,
    float* __restrict__ Aout)
{
  __shared__ h16 As[2][TILE_H];
  __shared__ h16 Bs[2][TILE_H];
  const int bh = blockIdx.z, b = bh / 12, hh = bh % 12;
  const h16* Aq = Q16 + (long)b * 1024 * 768 + hh * 64;
  const h16* Bk = K16 + (long)b * 512 * 768 + hh * 64;

  const int t = threadIdx.x, lane = t & 31, wave = t >> 5;
  const int m0 = (wave & 3) * 32, n0 = (wave >> 2) * 64;
  const long bm = (long)blockIdx.y * BM;
  const long bn = (long)blockIdx.x * BN;

  fv8 acc[2][4] = {};
  WMMA_TILE_LOOP(Aq, 768, Bk, 768, 64)

  const float scale = __expf(tau[0]) * 0.125f;  // exp(tau)/sqrt(64)
  float* out = Aout + (long)bh * 1024 * 512;
  const int rsub = (lane >> 4) * 8, csub = lane & 15;
#pragma unroll
  for (int im = 0; im < 2; ++im)
#pragma unroll
    for (int in = 0; in < 4; ++in) {
      const long row0 = bm + m0 + im * 16 + rsub;
      const long col  = bn + n0 + in * 16 + csub;
#pragma unroll
      for (int v = 0; v < 8; ++v) {
        const long r = row0 + v;
        const int ok = mask[(long)b * 1024 + r];
        out[r * 512 + col] = ok ? acc[im][in][v] * scale : -10000.0f;
      }
    }
}

// ---------------------------------------------------------------------------
// Row softmax over 512 elements (in place). One block per row, 2 elems/thread.
// Masked rows (all -10000) come out uniform 1/512, matching the reference.
// ---------------------------------------------------------------------------
__global__ __launch_bounds__(256) void softmax512(float* __restrict__ A)
{
  __shared__ float red[256];
  float* p = A + (long)blockIdx.x * 512;
  const int t = threadIdx.x;
  const float x0 = p[t], x1 = p[t + 256];
  red[t] = fmaxf(x0, x1);
  __syncthreads();
  for (int s = 128; s > 0; s >>= 1) {
    if (t < s) red[t] = fmaxf(red[t], red[t + s]);
    __syncthreads();
  }
  const float mx = red[0];
  __syncthreads();
  const float e0 = __expf(x0 - mx), e1 = __expf(x1 - mx);
  red[t] = e0 + e1;
  __syncthreads();
  for (int s = 128; s > 0; s >>= 1) {
    if (t < s) red[t] += red[t + s];
    __syncthreads();
  }
  const float inv = 1.0f / red[0];
  p[t] = e0 * inv;
  p[t + 256] = e1 * inv;
}

// ---------------------------------------------------------------------------
// Attention context: per (b,h): ctx[1024,64] = P(f32->f16) @ V_bh[512,64].
// V is [K,N] in memory -> transposed into the [N][K] LDS tile.
// grid = (1024/128, B*H), block = 256.
// ---------------------------------------------------------------------------
__global__ __launch_bounds__(256) void attn_ctx(
    const float* __restrict__ Aprob, const h16* __restrict__ V16,
    h16* __restrict__ ctx16)
{
  __shared__ h16 Asl[BM * LSA];
  __shared__ h16 Bsl[64 * LSA];
  const int bh = blockIdx.y, b = bh / 12, hh = bh % 12;
  const float* Ap = Aprob + (long)bh * 1024 * 512;
  const int t = threadIdx.x, lane = t & 31, wave = t >> 5;
  const int m0 = (wave & 3) * 32, n0 = (wave >> 2) * 32;
  const long bm = (long)blockIdx.x * BM;

  fv8 acc[2][2] = {};
  for (int k0 = 0; k0 < 512; k0 += BK) {
    // A tile: convert f32 probs -> f16
    int slot = t;
#pragma unroll
    for (int it = 0; it < 2; ++it, slot += 256) {
      const int row = slot >> 2, seg = (slot & 3) * 8;
      const float* src = Ap + (bm + row) * 512L + k0 + seg;
      const fv4 x0 = *(const fv4*)src;
      const fv4 x1 = *(const fv4*)(src + 4);
      h16* d = Asl + row * LSA + seg;
#pragma unroll
      for (int i = 0; i < 4; ++i) { d[i] = (h16)x0[i]; d[4 + i] = (h16)x1[i]; }
    }
    if (k0 + BK < 512)
      __builtin_prefetch(Ap + (bm + (t >> 2)) * 512L + k0 + BK, 0, 1);
    // B tile: V rows are K -> transpose into Bs[n][k]
    {
      const int k = t >> 3, nseg = (t & 7) * 8;
      const hv8 vv = *(const hv8*)(V16 + ((long)b * 512 + k0 + k) * 768L +
                                   hh * 64 + nseg);
#pragma unroll
      for (int i = 0; i < 8; ++i) Bsl[(nseg + i) * LSA + k] = vv[i];
    }
    __syncthreads();
    const hv16 a0 = frag_a(Asl, m0, lane);
    const hv16 a1 = frag_a(Asl, m0 + 16, lane);
    const hv16 b0 = frag_b(Bsl, n0, lane);
    const hv16 b1 = frag_b(Bsl, n0 + 16, lane);
    acc[0][0] = wmma16(a0, b0, acc[0][0]);
    acc[0][1] = wmma16(a0, b1, acc[0][1]);
    acc[1][0] = wmma16(a1, b0, acc[1][0]);
    acc[1][1] = wmma16(a1, b1, acc[1][1]);
    __syncthreads();
  }

  const int rsub = (lane >> 4) * 8, csub = lane & 15;
#pragma unroll
  for (int im = 0; im < 2; ++im)
#pragma unroll
    for (int in = 0; in < 2; ++in) {
      const long row0 = bm + m0 + im * 16 + rsub;
      const int  col  = n0 + in * 16 + csub;  // 0..63 within head
#pragma unroll
      for (int v = 0; v < 8; ++v)
        ctx16[((long)b * 1024 + row0 + v) * 768L + hh * 64 + col] =
            (h16)acc[im][in][v];
    }
}

// ---------------------------------------------------------------------------
// LayerNorm over 768 (3 elems/thread), writes final output.
// ---------------------------------------------------------------------------
__global__ __launch_bounds__(256) void layernorm768(
    const float* __restrict__ Y, const float* __restrict__ gamma,
    const float* __restrict__ beta, float* __restrict__ out)
{
  __shared__ float red[256];
  const long row = blockIdx.x;
  const float* y = Y + row * 768;
  const int t = threadIdx.x;
  const float a = y[t], b = y[t + 256], c = y[t + 512];
  red[t] = a + b + c;
  __syncthreads();
  for (int s = 128; s > 0; s >>= 1) {
    if (t < s) red[t] += red[t + s];
    __syncthreads();
  }
  const float mu = red[0] * (1.0f / 768.0f);
  __syncthreads();
  const float da = a - mu, db = b - mu, dc = c - mu;
  red[t] = da * da + db * db + dc * dc;
  __syncthreads();
  for (int s = 128; s > 0; s >>= 1) {
    if (t < s) red[t] += red[t + s];
    __syncthreads();
  }
  const float rs = rsqrtf(red[0] * (1.0f / 768.0f) + 1e-5f);
  out[row * 768 + t]       = da * rs * gamma[t]       + beta[t];
  out[row * 768 + t + 256] = db * rs * gamma[t + 256] + beta[t + 256];
  out[row * 768 + t + 512] = dc * rs * gamma[t + 512] + beta[t + 512];
}

// ---------------------------------------------------------------------------
__global__ void cvt_f32_f16(const float* __restrict__ s, h16* __restrict__ d,
                            long n)
{
  for (long i = (long)blockIdx.x * blockDim.x + threadIdx.x; i < n;
       i += (long)gridDim.x * blockDim.x)
    d[i] = (h16)s[i];
}

// ---------------------------------------------------------------------------
extern "C" void kernel_launch(void* const* d_in, const int* in_sizes, int n_in,
                              void* d_out, int out_size, void* d_ws,
                              size_t ws_size, hipStream_t stream)
{
  const float* q    = (const float*)d_in[0];
  const int*   mask = (const int*)d_in[1];
  const float* hla  = (const float*)d_in[2];
  const float* Wq   = (const float*)d_in[3];
  const float* Wk   = (const float*)d_in[4];
  const float* Wv   = (const float*)d_in[5];
  const float* Wout = (const float*)d_in[6];
  const float* bout = (const float*)d_in[7];
  const float* gamma= (const float*)d_in[8];
  const float* beta = (const float*)d_in[9];
  const float* W1   = (const float*)d_in[10];
  const float* b1   = (const float*)d_in[11];
  const float* W2   = (const float*)d_in[12];
  const float* b2   = (const float*)d_in[13];
  const float* tau  = (const float*)d_in[14];

  const long MQ = 16384, MK = 8192, D = 768, F = 3072;

  char* ws = (char*)d_ws;
  size_t off = 0;
  auto take = [&](size_t bytes) -> char* {
    char* p = ws + off;
    off += (bytes + 255) & ~(size_t)255;
    return p;
  };
  h16*  q16   = (h16*)take(MQ * D * 2);
  h16*  hla16 = (h16*)take(MK * D * 2);
  h16*  Wq16  = (h16*)take(D * D * 2);
  h16*  Wk16  = (h16*)take(D * D * 2);
  h16*  Wv16  = (h16*)take(D * D * 2);
  h16*  Wo16  = (h16*)take(D * D * 2);
  h16*  W116  = (h16*)take(F * D * 2);
  h16*  W216  = (h16*)take(D * F * 2);
  h16*  Q16   = (h16*)take(MQ * D * 2);
  h16*  K16   = (h16*)take(MK * D * 2);
  h16*  V16   = (h16*)take(MK * D * 2);
  h16*  ctx16 = (h16*)take(MQ * D * 2);
  float* qr   = (float*)take(MQ * D * 4);
  h16*  qr16  = (h16*)take(MQ * D * 2);
  h16*  h116  = (h16*)take(MQ * F * 2);
  float* yb   = (float*)take(MQ * D * 4);

  float* outLN = (float*)d_out;
  float* outA  = outLN + (size_t)MQ * D;  // A region: [16,12,1024,512] f32

  auto cvt = [&](const float* s, h16* d, long n) {
    cvt_f32_f16<<<2048, 256, 0, stream>>>(s, d, n);
  };
  cvt(q,    q16,   MQ * D);
  cvt(hla,  hla16, MK * D);
  cvt(Wq,   Wq16,  D * D);
  cvt(Wk,   Wk16,  D * D);
  cvt(Wv,   Wv16,  D * D);
  cvt(Wout, Wo16,  D * D);
  cvt(W1,   W116,  F * D);
  cvt(W2,   W216,  D * F);

  // Q/K/V projections (f16 out only)
  gemm_nt_f16<<<dim3(D / BN, MQ / BM), 256, 0, stream>>>(
      q16, D, Wq16, D, nullptr, Q16, D, nullptr, nullptr, D, 0);
  gemm_nt_f16<<<dim3(D / BN, MK / BM), 256, 0, stream>>>(
      hla16, D, Wk16, D, nullptr, K16, D, nullptr, nullptr, D, 0);
  gemm_nt_f16<<<dim3(D / BN, MK / BM), 256, 0, stream>>>(
      hla16, D, Wv16, D, nullptr, V16, D, nullptr, nullptr, D, 0);

  // Attention
  attn_logits<<<dim3(512 / BN, 1024 / BM, 16 * 12), 256, 0, stream>>>(
      Q16, K16, mask, tau, outA);
  softmax512<<<16 * 12 * 1024, 256, 0, stream>>>(outA);
  attn_ctx<<<dim3(1024 / BM, 16 * 12), 256, 0, stream>>>(outA, V16, ctx16);

  // Output projection + residual -> qr (f32 + f16)
  gemm_nt_f16<<<dim3(D / BN, MQ / BM), 256, 0, stream>>>(
      ctx16, D, Wo16, D, qr, qr16, D, bout, q, D, 0);
  // FFN up + exact GELU -> h1 (f16 only)
  gemm_nt_f16<<<dim3(F / BN, MQ / BM), 256, 0, stream>>>(
      qr16, D, W116, D, nullptr, h116, F, b1, nullptr, D, 1);
  // FFN down + bias + residual -> y (f32)
  gemm_nt_f16<<<dim3(D / BN, MQ / BM), 256, 0, stream>>>(
      h116, F, W216, F, yb, nullptr, D, b2, qr, F, 0);

  // LayerNorm -> out
  layernorm768<<<MQ, 256, 0, stream>>>(yb, gamma, beta, outLN);
}
